// DirectHODLR_61830349193987
// MI455X (gfx1250) — matrix operand order repaired and checked by reading, not compiled, not run on hardware
//
#include <hip/hip_runtime.h>

// ---------------- problem constants ----------------
#define B_DIM   64
#define N_DIM   262144      // 2^18
#define DEPTH   8
#define RANK    64
#define CHUNK   1024        // columns per workgroup; <= min block size (1024)
#define NCHUNK  (N_DIM / CHUNK)   // 256
#define SEC     128         // columns per LDS section
#define NSEC    (CHUNK / SEC)     // 8

// LDS pitches (in halves) — padded for alignment / bank spread
#define XPITCH  136         // x tile rows [b][c]
#define UPITCH  136         // transposed u rows [r][c] (phase 1)
#define UPITCH2 72          // natural u rows [c][r]   (phase 2)

typedef __attribute__((ext_vector_type(16))) _Float16 v16h;
typedef __attribute__((ext_vector_type(8)))  _Float16 v8h;
typedef __attribute__((ext_vector_type(4)))  _Float16 v4h;
typedef __attribute__((ext_vector_type(8)))  float    v8f;
typedef __attribute__((ext_vector_type(4)))  float    v4f;

static __device__ __forceinline__ v16h cat8(v8h lo, v8h hi) {
  return __builtin_shufflevector(lo, hi, 0,1,2,3,4,5,6,7,8,9,10,11,12,13,14,15);
}

// =====================================================================
// Phase 1: per-chunk partial projections  Tpart[chunk][lvl][b][r]
//   Tpart = sum over chunk columns c of x[b][c] * u[lvl][c][r]
// GEMM mapping: A = X (M=batch, K=c), B[k=c][n=r] = u[c][r]  (u transposed
// into LDS so B per-lane reads are 16 consecutive K halves).
// u staging software-pipelined: global loads for level l+1 are pinned
// (sched_barrier) to issue before the WMMA work of level l.
// =====================================================================
__global__ __launch_bounds__(256) void hodlr_phase1(
    const float* __restrict__ x, const float* __restrict__ u,
    float* __restrict__ Tpart)
{
  __shared__ alignas(16) _Float16 xh [64 * XPITCH];
  __shared__ alignas(16) _Float16 uth[2][RANK * UPITCH];

  const int tid  = threadIdx.x;
  const int wave = tid >> 5;
  const int lane = tid & 31;
  const int l16  = lane & 15;
  const int hi   = lane >> 4;          // 0: lanes 0-15, 1: lanes 16-31
  const int chunk = blockIdx.x;
  const unsigned n0 = chunk * CHUNK;

  const int m  = wave & 3;             // batch tile (16 rows each)
  const int n2 = wave >> 2;            // rank tile pair -> tiles 2*n2, 2*n2+1

  v8f acc[DEPTH][2];
  #pragma unroll
  for (int i = 0; i < DEPTH; ++i) { acc[i][0] = {}; acc[i][1] = {}; }

  for (int sec = 0; sec < NSEC; ++sec) {
    const unsigned cbase = n0 + sec * SEC;
    __syncthreads();                               // xh reuse fence
    // ---- stage x section (f32 -> f16), layout [b][c]: batch loads, then store
    {
      v4f xt[8];
      #pragma unroll
      for (int rnd = 0; rnd < 8; ++rnd) {
        unsigned b  = rnd * 8 + (tid >> 5);
        unsigned c4 = (tid & 31) * 4;
        xt[rnd] = *reinterpret_cast<const v4f*>(x + (b * (unsigned)N_DIM + cbase + c4));
      }
      #pragma unroll
      for (int rnd = 0; rnd < 8; ++rnd) {
        int b  = rnd * 8 + (tid >> 5);
        int c4 = (tid & 31) * 4;
        v4h hv = { (_Float16)xt[rnd].x, (_Float16)xt[rnd].y,
                   (_Float16)xt[rnd].z, (_Float16)xt[rnd].w };
        *reinterpret_cast<v4h*>(&xh[b * XPITCH + c4]) = hv;
      }
    }
    // ---- u pipeline registers + loader (8 x v4f = 128B in flight per thread)
    v4f ut[8];
    auto loadU = [&](int lvl) {
      const float* up = u + (size_t)lvl * ((size_t)N_DIM * RANK);  // uniform base
      #pragma unroll
      for (int rnd = 0; rnd < 8; ++rnd) {
        int idx = rnd * 256 + tid;
        unsigned c   = idx >> 4;
        unsigned r4  = (idx & 15) * 4;
        unsigned off = (cbase + c) * RANK + r4;    // 32-bit voffset, saddr form
        ut[rnd] = *reinterpret_cast<const v4f*>(up + off);
      }
    };
    loadU(0);
    #pragma unroll
    for (int lvl = 0; lvl < DEPTH; ++lvl) {
      const int buf = lvl & 1;
      // ---- store staged regs TRANSPOSED: uth[r][c] = u[lvl][cbase+c][r]
      #pragma unroll
      for (int rnd = 0; rnd < 8; ++rnd) {
        int idx = rnd * 256 + tid;
        int c   = idx >> 4;
        int r4  = (idx & 15) * 4;
        uth[buf][(r4 + 0) * UPITCH + c] = (_Float16)ut[rnd].x;
        uth[buf][(r4 + 1) * UPITCH + c] = (_Float16)ut[rnd].y;
        uth[buf][(r4 + 2) * UPITCH + c] = (_Float16)ut[rnd].z;
        uth[buf][(r4 + 3) * UPITCH + c] = (_Float16)ut[rnd].w;
      }
      __syncthreads();                             // staging visible
      if (lvl + 1 < DEPTH) {
        loadU(lvl + 1);                            // prefetch next level's u
        __builtin_amdgcn_sched_barrier(0);         // pin: loads issue BEFORE wmma
      }
      // ---- WMMA over this section's K range
      #pragma unroll
      for (int ks = 0; ks < SEC / 32; ++ks) {
        const _Float16* ap = &xh[(16 * m + l16) * XPITCH + ks * 32 + hi * 16];
        v16h A = cat8(*reinterpret_cast<const v8h*>(ap),
                      *reinterpret_cast<const v8h*>(ap + 8));
        const _Float16* bp0 = &uth[buf][(16 * (2 * n2 + 0) + l16) * UPITCH + ks * 32 + hi * 16];
        const _Float16* bp1 = &uth[buf][(16 * (2 * n2 + 1) + l16) * UPITCH + ks * 32 + hi * 16];
        v16h B0 = cat8(*reinterpret_cast<const v8h*>(bp0),
                       *reinterpret_cast<const v8h*>(bp0 + 8));
        v16h B1 = cat8(*reinterpret_cast<const v8h*>(bp1),
                       *reinterpret_cast<const v8h*>(bp1 + 8));
        acc[lvl][0] = __builtin_amdgcn_wmma_f32_16x16x32_f16(
            false, A, false, B0, (short)0, acc[lvl][0], false, false);
        acc[lvl][1] = __builtin_amdgcn_wmma_f32_16x16x32_f16(
            false, A, false, B1, (short)0, acc[lvl][1], false, false);
      }
    }
  }
  // ---- write per-chunk partials (f32): Tpart[(chunk*DEPTH+lvl)*4096 + b*64 + r]
  #pragma unroll
  for (int lvl = 0; lvl < DEPTH; ++lvl) {
    #pragma unroll
    for (int t = 0; t < 2; ++t) {
      unsigned r = 16 * (2 * n2 + t) + l16;
      #pragma unroll
      for (int j = 0; j < 8; ++j) {
        unsigned b = 16 * m + j + 8 * hi;          // C-layout: VGPR j -> M=j / j+8
        unsigned off = ((unsigned)chunk * DEPTH + lvl) * 4096u + b * 64u + r;
        Tpart[off] = acc[lvl][t][j];
      }
    }
  }
}

// =====================================================================
// Phase 1b: reduce chunk partials per (level, block), apply sibling swap,
// store f16 Tsib pre-formatted as row-major [b][r] for phase-2 A operands.
// Fixed summation order (2-way interleave) -> bitwise deterministic.
// =====================================================================
__global__ __launch_bounds__(256) void hodlr_reduce(
    const float* __restrict__ Tpart, _Float16* __restrict__ Tsib)
{
  const int bid = blockIdx.x;                      // 0..509 over all levels
  const int lvl = 30 - __clz(bid + 2);             // level from cumulative base
  const int base = (2 << lvl) - 2;                 // 2^(lvl+1) - 2
  const int p    = bid - base;                     // block index at this level
  const int nch  = 1 << (7 - lvl);                 // chunks per block
  const int chunk0 = p * nch;
  const int tid = threadIdx.x;

  #pragma unroll
  for (int k = 0; k < 4; ++k) {
    unsigned e4 = (k * 256 + tid) * 4;             // element b*64+r (vec4)
    v4f s0 = {}, s1 = {};
    int c = 0;
    for (; c + 1 < nch; c += 2) {
      s0 += *reinterpret_cast<const v4f*>(
          &Tpart[((unsigned)(chunk0 + c) * DEPTH + lvl) * 4096u + e4]);
      s1 += *reinterpret_cast<const v4f*>(
          &Tpart[((unsigned)(chunk0 + c + 1) * DEPTH + lvl) * 4096u + e4]);
    }
    if (c < nch)
      s0 += *reinterpret_cast<const v4f*>(
          &Tpart[((unsigned)(chunk0 + c) * DEPTH + lvl) * 4096u + e4]);
    v4f s = s0 + s1;
    // store T of block p into sibling slot p^1 => Tsib[slot] = T[slot^1]
    v4h hv = { (_Float16)s.x, (_Float16)s.y, (_Float16)s.z, (_Float16)s.w };
    *reinterpret_cast<v4h*>(&Tsib[(unsigned)(base + (p ^ 1)) * 4096u + e4]) = hv;
  }
}

// =====================================================================
// Phase 2: y[b][n] = diag[n]*x[b][n] + sum_lvl sum_r u[lvl][n][r]*Tsib[lvl][blk(n)][b][r]
// GEMM mapping: A = Tsib (M=batch, K=rank), B[k=r][n=c] = u[c][r]
// (u's natural [c][r] layout gives B directly; this code reads 16 consecutive ranks per lane).
// All 8 levels accumulated in registers -> y written exactly once.
// u staging software-pipelined with double-buffered LDS, as in phase 1.
// =====================================================================
__global__ __launch_bounds__(256) void hodlr_phase2(
    const float* __restrict__ x, const float* __restrict__ diag,
    const float* __restrict__ u, const _Float16* __restrict__ Tsib,
    float* __restrict__ y)
{
  __shared__ alignas(16) _Float16 uh[2][SEC * UPITCH2];

  const int tid  = threadIdx.x;
  const int wave = tid >> 5;
  const int lane = tid & 31;
  const int l16  = lane & 15;
  const int hi   = lane >> 4;
  const int chunk = blockIdx.x;
  const unsigned n0 = chunk * CHUNK;

  const int m  = wave & 3;                         // batch tile
  const int nq = (wave >> 2) * 4;                  // column tile group (0 or 4)

  for (int sec = 0; sec < NSEC; ++sec) {
    const unsigned cbase = n0 + sec * SEC;
    v8f acc[4];
    #pragma unroll
    for (int t = 0; t < 4; ++t) acc[t] = {};

    v4f ut[8];
    auto loadU = [&](int lvl) {
      const float* up = u + (size_t)lvl * ((size_t)N_DIM * RANK);  // uniform base
      #pragma unroll
      for (int rnd = 0; rnd < 8; ++rnd) {
        int idx = rnd * 256 + tid;
        unsigned c   = idx >> 4;
        unsigned r4  = (idx & 15) * 4;
        unsigned off = (cbase + c) * RANK + r4;    // 32-bit voffset, saddr form
        ut[rnd] = *reinterpret_cast<const v4f*>(up + off);
      }
    };
    loadU(0);
    #pragma unroll
    for (int lvl = 0; lvl < DEPTH; ++lvl) {
      const int buf = lvl & 1;
      // ---- store staged regs, natural layout uh[c][r]
      #pragma unroll
      for (int rnd = 0; rnd < 8; ++rnd) {
        int idx = rnd * 256 + tid;
        int c   = idx >> 4;
        int r4  = (idx & 15) * 4;
        v4h hv = { (_Float16)ut[rnd].x, (_Float16)ut[rnd].y,
                   (_Float16)ut[rnd].z, (_Float16)ut[rnd].w };
        *reinterpret_cast<v4h*>(&uh[buf][c * UPITCH2 + r4]) = hv;
      }
      __syncthreads();                             // staging visible
      if (lvl + 1 < DEPTH) {
        loadU(lvl + 1);                            // prefetch next level's u
        __builtin_amdgcn_sched_barrier(0);         // pin: loads issue BEFORE wmma
      }
      // ---- A operand: sibling-swapped T for this column's block (L2-resident)
      const unsigned p    = cbase >> (17 - lvl);   // block index (bs = 2^(17-lvl))
      const unsigned base = (2u << lvl) - 2u;
      const _Float16* Ab = Tsib + (size_t)(base + p) * 4096;
      #pragma unroll
      for (int ks = 0; ks < 2; ++ks) {             // K = 64 ranks, two k-steps
        const _Float16* ap = Ab + (16 * m + l16) * 64 + ks * 32 + hi * 16;
        v16h A = cat8(*reinterpret_cast<const v8h*>(ap),
                      *reinterpret_cast<const v8h*>(ap + 8));
        v16h Bv[4];
        #pragma unroll
        for (int t = 0; t < 4; ++t) {
          int c = 16 * (nq + t) + l16;
          const _Float16* bp = &uh[buf][c * UPITCH2 + ks * 32 + hi * 16];
          Bv[t] = cat8(*reinterpret_cast<const v8h*>(bp),
                       *reinterpret_cast<const v8h*>(bp + 8));
        }
        #pragma unroll
        for (int t = 0; t < 4; ++t) {
          acc[t] = __builtin_amdgcn_wmma_f32_16x16x32_f16(
              false, A, false, Bv[t], (short)0, acc[t], false, false);
        }
      }
    }
    // ---- finalize: y = diag*x + corrections  (f32, written once)
    #pragma unroll
    for (int t = 0; t < 4; ++t) {
      unsigned c = cbase + 16 * (nq + t) + l16;
      float dv = diag[c];
      #pragma unroll
      for (int j = 0; j < 8; ++j) {
        unsigned b = 16 * m + j + 8 * hi;
        unsigned off = b * (unsigned)N_DIM + c;
        y[off] = dv * x[off] + acc[t][j];
      }
    }
  }
}

// =====================================================================
extern "C" void kernel_launch(void* const* d_in, const int* in_sizes, int n_in,
                              void* d_out, int out_size, void* d_ws, size_t ws_size,
                              hipStream_t stream) {
  (void)in_sizes; (void)n_in; (void)out_size; (void)ws_size;
  const float* x    = (const float*)d_in[0];
  const float* diag = (const float*)d_in[1];
  const float* u    = (const float*)d_in[2];
  float* y = (float*)d_out;

  // workspace: Tpart f32 [NCHUNK][DEPTH][64*64]  (33.5 MB)
  //            Tsib  f16 [510 blocks][64*64]     ( 4.2 MB)
  float* Tpart = (float*)d_ws;
  _Float16* Tsib = (_Float16*)((char*)d_ws +
      (size_t)NCHUNK * DEPTH * 4096 * sizeof(float));

  hodlr_phase1<<<NCHUNK, 256, 0, stream>>>(x, u, Tpart);
  hodlr_reduce<<<510,    256, 0, stream>>>(Tpart, Tsib);
  hodlr_phase2<<<NCHUNK, 256, 0, stream>>>(x, diag, u, Tsib, y);
}
